// LSTMModule_30434138260129
// MI455X (gfx1250) — matrix operand order repaired and checked by reading, not compiled, not run on hardware
//
#include <hip/hip_runtime.h>
#include <hip/hip_bf16.h>

// Problem constants from the reference (fixed)
#define BB   4096
#define ZZ   512
#define HH   512
#define G4H  2048      // 4*H
#define NDD  10
#define TT   16

typedef __attribute__((ext_vector_type(16))) __bf16 v16bf;
typedef __attribute__((ext_vector_type(8)))  __bf16 v8bf;
typedef __attribute__((ext_vector_type(8)))  float  v8f;

// ---------------------------------------------------------------------------
// fp32 -> bf16 conversion (weights once, noise[t]/z per step)
// ---------------------------------------------------------------------------
__global__ __launch_bounds__(256) void cvt_f32_bf16(const float* __restrict__ src,
                                                    __bf16* __restrict__ dst, int n) {
    int i = blockIdx.x * blockDim.x + threadIdx.x;
    if (i < n) dst[i] = (__bf16)src[i];
}

// out[:,0,:] = z  AND  x_bf16 = bf16(z)
__global__ __launch_bounds__(256) void init_z_kernel(const float* __restrict__ z,
                                                     float* __restrict__ out,
                                                     __bf16* __restrict__ xb) {
    int i = blockIdx.x * blockDim.x + threadIdx.x;          // over B*Z
    if (i >= BB * ZZ) return;
    int b = i >> 9, zc = i & (ZZ - 1);
    out[(size_t)b * (TT * ZZ) + zc] = z[i];                  // slice t=0
    xb[i] = (__bf16)z[i];
}

// ---------------------------------------------------------------------------
// Fused gate GEMM:  gates[B,4H] = X[B,512]·Wihᵀ (+ H[B,512]·Whhᵀ) + bih + bhh
// bf16 WMMA, fp32 accumulate.
// Wave tile 64x64 (4x4 frags, 16 accumulators), block tile 256x128 (8 waves).
// ---------------------------------------------------------------------------
__device__ __forceinline__ v16bf load_afrag(const __bf16* __restrict__ base,
                                            int row, int k0, int halfSel) {
    // A (16x32, MxK): lanes 0-15 -> K {k0..k0+7, k0+16..k0+23}
    //                 lanes16-31 -> K {k0+8..k0+15, k0+24..k0+31}
    const __bf16* p = base + (size_t)row * ZZ;
    v8bf lo = *(const v8bf*)(p + k0 + halfSel * 8);
    v8bf hi = *(const v8bf*)(p + k0 + 16 + halfSel * 8);
    v16bf a;
#pragma unroll
    for (int i = 0; i < 8; ++i) { a[i] = lo[i]; a[i + 8] = hi[i]; }
    return a;
}

__device__ __forceinline__ v16bf load_bfrag(const __bf16* __restrict__ Wrow,
                                            int k0, int halfSel) {
    // B = Wᵀ (32x16, KxN): lane n holds W row n; lanes 0-15 -> K {k0..k0+7, k0+16..k0+23},
    //                      lanes 16-31 -> K {k0+8..k0+15, k0+24..k0+31}.
    v8bf lo = *(const v8bf*)(Wrow + k0 + halfSel * 8);
    v8bf hi = *(const v8bf*)(Wrow + k0 + 16 + halfSel * 8);
    v16bf b;
#pragma unroll
    for (int i = 0; i < 8; ++i) { b[i] = lo[i]; b[i + 8] = hi[i]; }
    return b;
}

__global__ __launch_bounds__(256)
void gates_gemm(const __bf16* __restrict__ X,    // [B,512] bf16
                const __bf16* __restrict__ Wih,  // [2048,512] bf16
                const __bf16* __restrict__ Hs,   // [B,512] bf16 (may be unused)
                const __bf16* __restrict__ Whh,  // [2048,512] bf16
                const float*  __restrict__ bih,
                const float*  __restrict__ bhh,
                float* __restrict__ gates,       // [B,2048] fp32
                int hasH) {
    const int lane    = threadIdx.x & 31;
    const int wave    = threadIdx.x >> 5;     // 0..7
    const int waveM   = wave & 3;             // 4 waves in M
    const int waveN   = wave >> 2;            // 2 waves in N
    const int laneM   = lane & 15;
    const int halfSel = lane >> 4;

    const int m0 = blockIdx.x * 256 + waveM * 64;   // wave M base (64 rows)
    const int n0 = blockIdx.y * 128 + waveN * 64;   // wave N base (64 cols)

    v8f acc[4][4] = {};

    // K loop: X · Wihᵀ
#pragma unroll 1
    for (int k0 = 0; k0 < ZZ; k0 += 32) {
        v16bf a[4], bf[4];
#pragma unroll
        for (int mf = 0; mf < 4; ++mf)
            a[mf] = load_afrag(X, m0 + mf * 16 + laneM, k0, halfSel);
#pragma unroll
        for (int nf = 0; nf < 4; ++nf)
            bf[nf] = load_bfrag(Wih + (size_t)(n0 + nf * 16 + laneM) * ZZ, k0, halfSel);
#pragma unroll
        for (int mf = 0; mf < 4; ++mf)
#pragma unroll
            for (int nf = 0; nf < 4; ++nf)
                acc[mf][nf] = __builtin_amdgcn_wmma_f32_16x16x32_bf16(
                    false, a[mf], false, bf[nf], (short)0, acc[mf][nf], false, false);
    }

    // K loop: H · Whhᵀ (recurrent term)
    if (hasH) {
#pragma unroll 1
        for (int k0 = 0; k0 < HH; k0 += 32) {
            v16bf a[4], bf[4];
#pragma unroll
            for (int mf = 0; mf < 4; ++mf)
                a[mf] = load_afrag(Hs, m0 + mf * 16 + laneM, k0, halfSel);
#pragma unroll
            for (int nf = 0; nf < 4; ++nf)
                bf[nf] = load_bfrag(Whh + (size_t)(n0 + nf * 16 + laneM) * HH, k0, halfSel);
#pragma unroll
            for (int mf = 0; mf < 4; ++mf)
#pragma unroll
                for (int nf = 0; nf < 4; ++nf)
                    acc[mf][nf] = __builtin_amdgcn_wmma_f32_16x16x32_bf16(
                        false, a[mf], false, bf[nf], (short)0, acc[mf][nf], false, false);
        }
    }

    // store with bias (bias depends only on column n)
#pragma unroll
    for (int nf = 0; nf < 4; ++nf) {
        int n = n0 + nf * 16 + laneM;
        float bias = bih[n] + bhh[n];
#pragma unroll
        for (int mf = 0; mf < 4; ++mf) {
            int rowBase = m0 + mf * 16 + halfSel * 8;
            float* gp = gates + (size_t)rowBase * G4H + n;
#pragma unroll
            for (int v = 0; v < 8; ++v)
                gp[(size_t)v * G4H] = acc[mf][nf][v] + bias;
        }
    }
}

// ---------------------------------------------------------------------------
// Pointwise LSTM cell update (gate order i,f,g,o)
// ---------------------------------------------------------------------------
__global__ __launch_bounds__(256)
void lstm_pointwise(const float* __restrict__ gates, float* __restrict__ c,
                    float* __restrict__ h, __bf16* __restrict__ hb, int first) {
    int idx = blockIdx.x * blockDim.x + threadIdx.x;     // over B*H
    if (idx >= BB * HH) return;
    int b = idx >> 9, j = idx & (HH - 1);
    const float* g = gates + (size_t)b * G4H;
    float gi = g[j], gf = g[HH + j], gg = g[2 * HH + j], go = g[3 * HH + j];
    float si = 1.f / (1.f + __expf(-gi));
    float sf = 1.f / (1.f + __expf(-gf));
    float so = 1.f / (1.f + __expf(-go));
    float cprev = first ? 0.f : c[idx];
    float cn = sf * cprev + si * tanhf(gg);
    float hn = so * tanhf(cn);
    c[idx] = cn;
    h[idx] = hn;
    hb[idx] = (__bf16)hn;
}

// ---------------------------------------------------------------------------
// Out projection: mul = tanh(h·w + b); out[:,t,:] = out[:,t-1,:] + 0.2*(mul·dir)
// One wave32 per batch row; 8 rows per block.
// ---------------------------------------------------------------------------
__global__ __launch_bounds__(256)
void out_proj(const float* __restrict__ h, const float* __restrict__ w,
              const float* __restrict__ bias, const float* __restrict__ dir,
              float* __restrict__ out, int t) {
    int lane = threadIdx.x & 31;
    int wave = threadIdx.x >> 5;
    int b = blockIdx.x * 8 + wave;
    if (b >= BB) return;

    const float* hr = h + (size_t)b * HH;
    float p[NDD];
#pragma unroll
    for (int d = 0; d < NDD; ++d) p[d] = 0.f;
    for (int k = lane; k < HH; k += 32) {
        float hv = hr[k];
        const float* wr = w + (size_t)k * NDD;
#pragma unroll
        for (int d = 0; d < NDD; ++d) p[d] += hv * wr[d];
    }
#pragma unroll
    for (int off = 16; off > 0; off >>= 1)
#pragma unroll
        for (int d = 0; d < NDD; ++d) p[d] += __shfl_xor(p[d], off, 32);

    float mul[NDD];
#pragma unroll
    for (int d = 0; d < NDD; ++d) mul[d] = tanhf(p[d] + bias[d]);

    float* orow = out + (size_t)b * (TT * ZZ) + (size_t)t * ZZ;
    const float* prow = orow - ZZ;
    for (int zc = lane; zc < ZZ; zc += 32) {
        float s = 0.f;
#pragma unroll
        for (int d = 0; d < NDD; ++d) s += mul[d] * dir[(size_t)d * ZZ + zc];
        orow[zc] = prow[zc] + 0.2f * s;
    }
}

// ---------------------------------------------------------------------------
// Host-side launch
// ---------------------------------------------------------------------------
extern "C" void kernel_launch(void* const* d_in, const int* in_sizes, int n_in,
                              void* d_out, int out_size, void* d_ws, size_t ws_size,
                              hipStream_t stream) {
    (void)in_sizes; (void)n_in; (void)out_size; (void)ws_size;
    const float* z       = (const float*)d_in[0];
    const float* noise   = (const float*)d_in[1];
    // d_in[2] = n_frame (== 16, fixed by the reference)
    const float* Wih_enc = (const float*)d_in[3];
    // d_in[4] = Whh_enc (unused: enc cell runs with h=0)
    const float* bih_enc = (const float*)d_in[5];
    const float* bhh_enc = (const float*)d_in[6];
    const float* Wih     = (const float*)d_in[7];
    const float* Whh     = (const float*)d_in[8];
    const float* bih     = (const float*)d_in[9];
    const float* bhh     = (const float*)d_in[10];
    const float* w       = (const float*)d_in[11];
    const float* bv      = (const float*)d_in[12];
    const float* fwd     = (const float*)d_in[13];
    const float* bwd     = (const float*)d_in[14];
    float* out = (float*)d_out;

    // Workspace layout (bytes)
    char* ws = (char*)d_ws;
    __bf16* WihB    = (__bf16*)(ws + 0);                       // 2 MB
    __bf16* WhhB    = (__bf16*)(ws + (2u  << 20));             // 2 MB
    __bf16* WencB   = (__bf16*)(ws + (4u  << 20));             // 2 MB
    __bf16* xB      = (__bf16*)(ws + (6u  << 20));             // 4 MB
    __bf16* hB      = (__bf16*)(ws + (10u << 20));             // 4 MB
    float*  gates   = (float*) (ws + (14u << 20));             // 32 MB
    float*  hF      = (float*) (ws + (46u << 20));             // 8 MB
    float*  cF      = (float*) (ws + (54u << 20));             // 8 MB

    const int nW  = G4H * ZZ;        // 2048*512
    const int nBZ = BB * ZZ;         // 4096*512
    dim3 blk(256);
    dim3 gridW((nW + 255) / 256), gridBZ((nBZ + 255) / 256);
    dim3 gridGemm(BB / 256, G4H / 128);     // 16 x 16 blocks
    dim3 gridProj(BB / 8);                  // 512 blocks, 8 rows each

    // 1) weights -> bf16
    cvt_f32_bf16<<<gridW, blk, 0, stream>>>(Wih,     WihB,  nW);
    cvt_f32_bf16<<<gridW, blk, 0, stream>>>(Whh,     WhhB,  nW);
    cvt_f32_bf16<<<gridW, blk, 0, stream>>>(Wih_enc, WencB, nW);

    // 2) out[:,0,:] = z ; xB = bf16(z)
    init_z_kernel<<<gridBZ, blk, 0, stream>>>(z, out, xB);

    // 3) encoder cell (h=0, c=0): gates = z·Wih_encᵀ + bih_enc + bhh_enc
    gates_gemm<<<gridGemm, blk, 0, stream>>>(xB, WencB, hB, WhhB,
                                             bih_enc, bhh_enc, gates, /*hasH=*/0);
    lstm_pointwise<<<gridBZ, blk, 0, stream>>>(gates, cF, hF, hB, /*first=*/1);

    // 4) 15 recurrent steps
    for (int i = 0; i < TT - 1; ++i) {
        cvt_f32_bf16<<<gridBZ, blk, 0, stream>>>(noise + (size_t)i * nBZ, xB, nBZ);
        gates_gemm<<<gridGemm, blk, 0, stream>>>(xB, WihB, hB, WhhB,
                                                 bih, bhh, gates, /*hasH=*/1);
        lstm_pointwise<<<gridBZ, blk, 0, stream>>>(gates, cF, hF, hB, /*first=*/0);
        const float* dir = (i % 2 == 0) ? bwd : fwd;
        out_proj<<<gridProj, blk, 0, stream>>>(hF, w, bv, dir, out, i + 1);
    }
}
